// FactorizedWorldModel_75591424410011
// MI455X (gfx1250) — matrix operand order repaired and checked by reading, not compiled, not run on hardware
//
#include <hip/hip_runtime.h>
#include <hip/hip_bf16.h>
#include <cstdint>
#include <cstddef>

// ---------------------------------------------------------------------------
// MI455X (gfx1250) implementation. All matrix math routed through
// v_wmma_f32_16x16x32_f16 (f16 inputs, f32 accumulate). wave32 everywhere.
// Big 5x5 convs use async global->LDS staging (ASYNCcnt) per CDNA5.
// ---------------------------------------------------------------------------

typedef __attribute__((ext_vector_type(16))) _Float16 v16h;
typedef __attribute__((ext_vector_type(8)))  float    v8f;
typedef __attribute__((ext_vector_type(8)))  unsigned v8u;

#define DEVI __device__ __forceinline__

DEVI unsigned pack2f(float lo, float hi) {
  _Float16 l = (_Float16)lo, h = (_Float16)hi;
  unsigned short lu = __builtin_bit_cast(unsigned short, l);
  unsigned short hu = __builtin_bit_cast(unsigned short, h);
  return (unsigned)lu | ((unsigned)hu << 16);
}

DEVI v8f wmma_f16(const v8u& a, const v8u& b, v8f c) {
  return __builtin_amdgcn_wmma_f32_16x16x32_f16(
      false, __builtin_bit_cast(v16h, a),
      false, __builtin_bit_cast(v16h, b),
      (short)0, c, false, false);
}

DEVI float sigmf(float x) { return 1.0f / (1.0f + expf(-x)); }
DEVI float gelu_tanh(float x) {
  return 0.5f * x * (1.0f + tanhf(0.7978845608f * (x + 0.044715f * x * x * x)));
}

// CDNA5 async global->LDS copy (16B), tracked by ASYNCcnt.
DEVI void async_load_b128(unsigned lds_off, const void* gaddr) {
  asm volatile("global_load_async_to_lds_b128 %0, %1, off"
               :: "v"(lds_off), "v"((unsigned long long)(uintptr_t)gaddr)
               : "memory");
}
DEVI void wait_asynccnt0() { asm volatile("s_wait_asynccnt 0" ::: "memory"); }

// ---------------------------------------------------------------------------
// pack_b: rearrange a K x N (or transposed N x K) f32 matrix into the CDNA5
// B-fragment dword order: out[((c*ntiles + t)*32 + lane)*8 + r] packs
// (K = c*32 + (lane<16?0:8) + (r<4 ? 2r : 16 + 2(r-4)) [,+1], N = t*16 + (lane&15)). Zero-pads.
// ---------------------------------------------------------------------------
__global__ __launch_bounds__(32)
void pack_b(const float* __restrict__ src, long sstride, int Kd, int N, int trans,
            unsigned* __restrict__ out, long ostride) {
  int c = blockIdx.x, t = blockIdx.y, b = blockIdx.z, lane = threadIdx.x;
  src += (long)b * sstride;
  out += (long)b * ostride;
  int n  = t * 16 + (lane & 15);
  int kb = c * 32 + ((lane >> 4) << 3);
  unsigned* o = out + (((long)c * gridDim.y + t) * 32 + lane) * 8;
#pragma unroll
  for (int r = 0; r < 8; r++) {
    int k0 = kb + ((r < 4) ? 2 * r : 16 + 2 * (r - 4)), k1 = k0 + 1;
    float lo = (n < N && k0 < Kd) ? (trans ? src[(long)n * Kd + k0] : src[(long)k0 * N + n]) : 0.f;
    float hi = (n < N && k1 < Kd) ? (trans ? src[(long)n * Kd + k1] : src[(long)k1 * N + n]) : 0.f;
    o[r] = pack2f(lo, hi);
  }
}

// ---------------------------------------------------------------------------
// Generic WMMA GEMM: C[M,N] = act(alpha*(A@B) + bias + add). One 16x16 tile
// per wave. A: f16 row-major (lda = padded K, multiple of 32, zero-padded).
// Row index clamped (not predicated) for the M-tail: stored rows only depend
// on their own A row, so no exec-mask churn in the K-loop.
// ---------------------------------------------------------------------------
__global__ __launch_bounds__(32)
void wmma_gemm(const _Float16* __restrict__ A, int lda, long sA,
               const unsigned* __restrict__ Bp, long sB,
               const float* __restrict__ bias,
               const float* __restrict__ add, long sAdd,
               float* __restrict__ Cf, long sCf,
               _Float16* __restrict__ Ch, int ldch, long sCh,
               int M, int N, int Kd, int act, float alpha) {
  int lane = threadIdx.x;
  int mt = blockIdx.x, nt = blockIdx.y, b = blockIdx.z;
  const _Float16* Ab = A + (long)b * sA;
  const unsigned* Bb = Bp + (long)b * sB;
  int row = mt * 16 + (lane & 15);
  if (row >= M) row = M - 1;  // clamp: harmless, stores are predicated
  int kh = (lane >> 4) << 3;
  const unsigned* ar = (const unsigned*)(Ab + (long)row * lda);
  int chunks = (Kd + 31) >> 5;
  long bstep = (long)gridDim.y * 256;
  const unsigned* bp = Bb + (long)nt * 256 + lane * 8;
  v8f acc = {};
  for (int c = 0; c < chunks; c++) {
    int o = (c << 4) + (kh >> 1);
    v8u au;
    au[0] = ar[o];     au[1] = ar[o + 1]; au[2] = ar[o + 2];  au[3] = ar[o + 3];
    au[4] = ar[o + 8]; au[5] = ar[o + 9]; au[6] = ar[o + 10]; au[7] = ar[o + 11];
    __builtin_prefetch(ar + o + 16, 0, 1);  // global_prefetch_b8
    v8u bu;
    bu[0] = bp[0]; bu[1] = bp[1]; bu[2] = bp[2]; bu[3] = bp[3];
    bu[4] = bp[4]; bu[5] = bp[5]; bu[6] = bp[6]; bu[7] = bp[7];
    bp += bstep;
    acc = wmma_f16(au, bu, acc);
  }
  int col = nt * 16 + (lane & 15);
  if (col >= N) return;
  int rbase = mt * 16 + ((lane >> 4) << 3);
  float bs = bias ? bias[col] : 0.f;
#pragma unroll
  for (int r = 0; r < 8; r++) {
    int rr = rbase + r;
    if (rr >= M) continue;
    float v = acc[r] * alpha + bs;
    if (add) v += add[(long)b * sAdd + (long)rr * N + col];
    if (act == 1) v = fmaxf(v, 0.f);
    else if (act == 2) v = gelu_tanh(v);
    if (Cf) Cf[(long)b * sCf + (long)rr * N + col] = v;
    if (Ch) Ch[(long)b * sCh + (long)rr * ldch + col] = (_Float16)v;
  }
}

// ---------------------------------------------------------------------------
// Implicit-GEMM SAME conv (NHWC, f16 in), direct-from-global gather. Used for
// the Cin=3 first conv and the 3x3 Cout=4 head only.
// ---------------------------------------------------------------------------
template <int NCT>
__global__ __launch_bounds__(32)
void conv_wmma(const _Float16* __restrict__ X, const unsigned* __restrict__ Wp,
               const float* __restrict__ bias, float* __restrict__ Yf,
               _Float16* __restrict__ Yh, int H, int W, int Cin, int Cout,
               int ksz, int relu) {
  int lane = threadIdx.x;
  int wt = W >> 4;
  int tb = blockIdx.x;
  int xt = tb % wt; int y = (tb / wt) % H; int n = tb / (wt * H);
  int x = (xt << 4) + (lane & 15);
  int kh = (lane >> 4) << 3;
  int P = ksz >> 1;
  int Ktot = ksz * ksz * Cin;
  int chunks = (Ktot + 31) >> 5;
  v8f acc[NCT] = {};
  const unsigned* bp = Wp + (long)lane * 8;
  for (int c = 0; c < chunks; c++) {
    v8u au = {};
    if ((Cin & 31) == 0) {
      int cpt = Cin >> 5;
      int tap = c / cpt; int cib = (c - tap * cpt) << 5;
      int ky = tap / ksz, kx = tap - ky * ksz;
      int iy = y + ky - P, ix = x + kx - P;
      if (iy >= 0 && iy < H && ix >= 0 && ix < W) {
        const unsigned* p = (const unsigned*)(X + ((long)(n * H + iy) * W + ix) * Cin + cib);
        int o = kh >> 1;
        au[0] = p[o];     au[1] = p[o + 1]; au[2] = p[o + 2];  au[3] = p[o + 3];
        au[4] = p[o + 8]; au[5] = p[o + 9]; au[6] = p[o + 10]; au[7] = p[o + 11];
      }
    } else {
      unsigned short hv[16];
#pragma unroll
      for (int e = 0; e < 16; e++) {
        int koff = (e < 8) ? (kh + e) : (8 + kh + e);
        int k = (c << 5) + koff;
        _Float16 val = (_Float16)0.f;
        if (k < Ktot) {
          int tap = k / Cin; int ci = k - tap * Cin;
          int ky = tap / ksz, kx = tap - ky * ksz;
          int iy = y + ky - P, ix = x + kx - P;
          if (iy >= 0 && iy < H && ix >= 0 && ix < W)
            val = X[((long)(n * H + iy) * W + ix) * Cin + ci];
        }
        hv[e] = __builtin_bit_cast(unsigned short, val);
      }
#pragma unroll
      for (int e = 0; e < 8; e++)
        au[e] = (unsigned)hv[2 * e] | ((unsigned)hv[2 * e + 1] << 16);
    }
#pragma unroll
    for (int j = 0; j < NCT; j++) {
      const unsigned* q = bp + (long)(c * NCT + j) * 256;
      v8u bu;
      bu[0] = q[0]; bu[1] = q[1]; bu[2] = q[2]; bu[3] = q[3];
      bu[4] = q[4]; bu[5] = q[5]; bu[6] = q[6]; bu[7] = q[7];
      acc[j] = wmma_f16(au, bu, acc[j]);
    }
  }
  int cl = lane & 15;
  int xb = (xt << 4) + ((lane >> 4) << 3);
#pragma unroll
  for (int j = 0; j < NCT; j++) {
    int co = j * 16 + cl;
    if (co >= Cout) continue;
    float bs = bias ? bias[co] : 0.f;
#pragma unroll
    for (int r = 0; r < 8; r++) {
      int xs = xb + r;
      float v = acc[j][r] + bs;
      if (relu) v = fmaxf(v, 0.f);
      long idx = ((long)(n * H + y) * W + xs) * Cout + co;
      if (Yf) Yf[idx] = v;
      if (Yh) Yh[idx] = (_Float16)v;
    }
  }
}

// ---------------------------------------------------------------------------
// LDS-staged 5x5 SAME conv, Cin=Cout=64, W=H=64. One workgroup (4 waves) per
// output row: async-copy the 5-row x 68-col x 64-ch halo strip (43.5 KB) into
// LDS (global_load_async_to_lds_b128 / s_wait_asynccnt), then each wave
// computes 16 pixels x 64 couts from LDS with no boundary predication.
// ---------------------------------------------------------------------------
__global__ __launch_bounds__(128)
void conv5_lds(const _Float16* __restrict__ X, const unsigned* __restrict__ Wp,
               const float* __restrict__ bias, float* __restrict__ Yf,
               _Float16* __restrict__ Yh, int relu) {
  constexpr int H = 64, W = 64, C = 64, WCOL = 68;  // 2-px halo each side
  __shared__ _Float16 smem[5 * WCOL * C];           // 43520 B
  int tid = threadIdx.x;
  int n = blockIdx.x >> 6;
  int y = blockIdx.x & 63;

  // ---- stage input strip: rows y-2..y+2, cols -2..65 (zeros outside) ------
  unsigned* s32 = (unsigned*)smem;
  for (int i = tid; i < 5 * WCOL * 8; i += 128) {      // 16B chunks
    int r = i / (WCOL * 8);
    int rem = i - r * (WCOL * 8);
    int j = rem >> 3;            // LDS column 0..67
    int g = rem & 7;             // 8-half group within channel dim
    int iy = y + r - 2;
    int hoff = ((r * WCOL) + j) * C + g * 8;           // halves
    if (iy >= 0 && iy < H && j >= 2 && j <= 65) {
      const _Float16* gp = X + ((long)(n * H + iy) * W + (j - 2)) * C + g * 8;
      async_load_b128((unsigned)(hoff * 2), gp);
    } else {
      int o = hoff >> 1;
      s32[o] = 0; s32[o + 1] = 0; s32[o + 2] = 0; s32[o + 3] = 0;
    }
  }
  wait_asynccnt0();
  __syncthreads();

  // ---- compute: wave w -> pixels [w*16, w*16+16) ---------------------------
  int lane = tid & 31;
  int wid = tid >> 5;
  int x = (wid << 4) + (lane & 15);
  int kh = (lane >> 4) << 3;
  v8f acc[4] = {};
  const unsigned* bp = Wp + (long)lane * 8;
  for (int c = 0; c < 50; c++) {
    int tap = c >> 1; int cib = (c & 1) << 5;
    int ky = tap / 5, kx = tap - ky * 5;
    int col = x + kx;                                  // 0..67, always valid
    int o = ((((ky * WCOL) + col) * C + cib) >> 1) + (kh >> 1);
    v8u au;
    au[0] = s32[o];     au[1] = s32[o + 1]; au[2] = s32[o + 2];  au[3] = s32[o + 3];
    au[4] = s32[o + 8]; au[5] = s32[o + 9]; au[6] = s32[o + 10]; au[7] = s32[o + 11];
#pragma unroll
    for (int j = 0; j < 4; j++) {
      const unsigned* q = bp + (long)(c * 4 + j) * 256;
      v8u bu;
      bu[0] = q[0]; bu[1] = q[1]; bu[2] = q[2]; bu[3] = q[3];
      bu[4] = q[4]; bu[5] = q[5]; bu[6] = q[6]; bu[7] = q[7];
      acc[j] = wmma_f16(au, bu, acc[j]);
    }
  }
  int cl = lane & 15;
  int xb = (wid << 4) + ((lane >> 4) << 3);
#pragma unroll
  for (int j = 0; j < 4; j++) {
    int co = j * 16 + cl;
    float bs = bias[co];
#pragma unroll
    for (int r = 0; r < 8; r++) {
      float v = acc[j][r] + bs;
      if (relu) v = fmaxf(v, 0.f);
      long idx = ((long)(n * H + y) * W + (xb + r)) * C + co;
      if (Yf) Yf[idx] = v;
      if (Yh) Yh[idx] = (_Float16)v;
    }
  }
}

// ---------------------------------------------------------------------------
// Stride-2 conv_transpose 5x5 SAME (jax: pad_a=3, gather p=y+k-3, p even),
// kernel used un-flipped (transpose_kernel=False). Cin must be mult of 32.
// ---------------------------------------------------------------------------
template <int NCT>
__global__ __launch_bounds__(32)
void deconv2_wmma(const _Float16* __restrict__ X, const unsigned* __restrict__ Wp,
                  const float* __restrict__ bias, _Float16* __restrict__ Yh,
                  int Hin, int Win, int Cin, int Cout, int relu) {
  int Ho = Hin << 1, Wo = Win << 1;
  int lane = threadIdx.x;
  int wt = Wo >> 4;
  int tb = blockIdx.x;
  int xt = tb % wt; int y = (tb / wt) % Ho; int n = tb / (wt * Ho);
  int x = (xt << 4) + (lane & 15);
  int kh = (lane >> 4) << 3;
  int cpt = Cin >> 5;
  int chunks = 25 * cpt;
  v8f acc[NCT] = {};
  const unsigned* bp = Wp + (long)lane * 8;
  for (int c = 0; c < chunks; c++) {
    int tap = c / cpt; int cib = (c - tap * cpt) << 5;
    int ky = tap / 5, kx = tap - ky * 5;
    int py = y + ky - 3, px = x + kx - 3;
    v8u au = {};
    if (py >= 0 && !(py & 1) && (py >> 1) < Hin &&
        px >= 0 && !(px & 1) && (px >> 1) < Win) {
      const unsigned* p = (const unsigned*)(X + ((long)(n * Hin + (py >> 1)) * Win + (px >> 1)) * Cin + cib);
      int o = kh >> 1;
      au[0] = p[o];     au[1] = p[o + 1]; au[2] = p[o + 2];  au[3] = p[o + 3];
      au[4] = p[o + 8]; au[5] = p[o + 9]; au[6] = p[o + 10]; au[7] = p[o + 11];
    }
#pragma unroll
    for (int j = 0; j < NCT; j++) {
      const unsigned* q = bp + (long)(c * NCT + j) * 256;
      v8u bu;
      bu[0] = q[0]; bu[1] = q[1]; bu[2] = q[2]; bu[3] = q[3];
      bu[4] = q[4]; bu[5] = q[5]; bu[6] = q[6]; bu[7] = q[7];
      acc[j] = wmma_f16(au, bu, acc[j]);
    }
  }
  int cl = lane & 15;
  int xb = (xt << 4) + ((lane >> 4) << 3);
#pragma unroll
  for (int j = 0; j < NCT; j++) {
    int co = j * 16 + cl;
    if (co >= Cout) continue;
    float bs = bias ? bias[co] : 0.f;
#pragma unroll
    for (int r = 0; r < 8; r++) {
      int xs = xb + r;
      float v = acc[j][r] + bs;
      if (relu) v = fmaxf(v, 0.f);
      Yh[((long)(n * Ho + y) * Wo + xs) * Cout + co] = (_Float16)v;
    }
  }
}

// ---------------------------------------------------------------------------
// Pointwise / reduction helpers (wave32 LN over D=64, softmaxes, GRU, etc.)
// ---------------------------------------------------------------------------
__global__ __launch_bounds__(32)
void ln64(const float* __restrict__ X, long bstride, const float* __restrict__ sc,
          const float* __restrict__ bi, float* __restrict__ Yf,
          _Float16* __restrict__ Yh) {
  int lane = threadIdx.x;
  const float* p = X + (long)blockIdx.y * bstride + (long)blockIdx.x * 64;
  float x0 = p[lane], x1 = p[lane + 32];
  float s = x0 + x1, q = x0 * x0 + x1 * x1;
  for (int m = 16; m; m >>= 1) { s += __shfl_xor(s, m, 32); q += __shfl_xor(q, m, 32); }
  float mean = s * 0.015625f;
  float var = q * 0.015625f - mean * mean;
  float rs = rsqrtf(var + 1e-5f);
  long ro = ((long)blockIdx.y * gridDim.x + blockIdx.x) * 64;
  float y0 = (x0 - mean) * rs * sc[lane] + bi[lane];
  float y1 = (x1 - mean) * rs * sc[lane + 32] + bi[lane + 32];
  if (Yf) { Yf[ro + lane] = y0; Yf[ro + lane + 32] = y1; }
  if (Yh) { Yh[ro + lane] = (_Float16)y0; Yh[ro + lane + 32] = (_Float16)y1; }
}

__global__ void cvt_f32_f16(const float* __restrict__ x, _Float16* __restrict__ y, long n) {
  long i = (long)blockIdx.x * blockDim.x + threadIdx.x;
  if (i < n) y[i] = (_Float16)x[i];
}

__global__ void pad_cvt(const float* __restrict__ x, _Float16* __restrict__ y,
                        int rows, int Kd, int lda) {
  int i = blockIdx.x * blockDim.x + threadIdx.x;
  if (i >= rows * lda) return;
  int r = i / lda, c = i - r * lda;
  y[i] = (c < Kd) ? (_Float16)x[(long)r * Kd + c] : (_Float16)0.f;
}

__global__ void posadd_enc(float* __restrict__ x, const float* __restrict__ pw,
                           const float* __restrict__ pb, long total) {
  long i = (long)blockIdx.x * blockDim.x + threadIdx.x;
  if (i >= total) return;
  int d = (int)(i & 63);
  long p = (i >> 6) & 4095;
  int yy = (int)(p >> 6), xx = (int)(p & 63);
  float g = 1.f / 63.f, gy = yy * g, gx = xx * g;
  x[i] += gy * pw[d] + gx * pw[64 + d] + (1.f - gy) * pw[128 + d] + (1.f - gx) * pw[192 + d] + pb[d];
}

__global__ void colsoftmax5(const float* __restrict__ lg, float* __restrict__ at, int total) {
  int i = blockIdx.x * blockDim.x + threadIdx.x;
  if (i >= total) return;
  int b = i >> 12, n = i & 4095;
  const float* p = lg + (long)b * 5 * 4096 + n;
  float v[5]; float m = -1e30f;
#pragma unroll
  for (int k = 0; k < 5; k++) { v[k] = p[(long)k * 4096]; m = fmaxf(m, v[k]); }
  float s = 0.f;
#pragma unroll
  for (int k = 0; k < 5; k++) { v[k] = expf(v[k] - m); s += v[k]; }
  float* q = at + (long)b * 5 * 4096 + n;
#pragma unroll
  for (int k = 0; k < 5; k++) q[(long)k * 4096] = v[k] / s;
}

__global__ void rowsum(const float* __restrict__ x, float* __restrict__ out, int cols) {
  __shared__ float sm[256];
  int r = blockIdx.x;
  const float* p = x + (long)r * cols;
  float s = 0.f;
  for (int i = threadIdx.x; i < cols; i += blockDim.x) s += p[i];
  sm[threadIdx.x] = s; __syncthreads();
  for (int o = blockDim.x >> 1; o > 0; o >>= 1) {
    if ((int)threadIdx.x < o) sm[threadIdx.x] += sm[threadIdx.x + o];
    __syncthreads();
  }
  if (threadIdx.x == 0) out[r] = sm[0];
}

__global__ void divnorm(const float* __restrict__ at, const float* __restrict__ sums,
                        _Float16* __restrict__ o, int total) {
  int i = blockIdx.x * blockDim.x + threadIdx.x;
  if (i >= total) return;
  int row = i >> 12;
  o[i] = (_Float16)(at[i] / (sums[row] + 1e-8f));
}

__global__ void gru_fuse(const float* __restrict__ gx, const float* __restrict__ gh,
                         float* __restrict__ slots, _Float16* __restrict__ slots16, int total) {
  int i = blockIdx.x * blockDim.x + threadIdx.x;
  if (i >= total) return;
  int r = i >> 6, d = i & 63;
  const float* x = gx + (long)r * 192;
  const float* h = gh + (long)r * 192;
  float z  = sigmf(x[d] + h[d]);
  float rg = sigmf(x[64 + d] + h[64 + d]);
  float hh = tanhf(x[128 + d] + rg * h[128 + d]);
  float pv = slots[i];
  float o = (1.f - z) * hh + z * pv;
  slots[i] = o; slots16[i] = (_Float16)o;
}

__global__ void softmax6_cvt(const float* __restrict__ l, _Float16* __restrict__ o, int rows) {
  int r = blockIdx.x * blockDim.x + threadIdx.x;
  if (r >= rows) return;
  const float* p = l + (long)r * 6;
  float m = p[0];
#pragma unroll
  for (int i = 1; i < 6; i++) m = fmaxf(m, p[i]);
  float e[6], s = 0.f;
#pragma unroll
  for (int i = 0; i < 6; i++) { e[i] = expf(p[i] - m); s += e[i]; }
  _Float16* q = o + (long)r * 32;
#pragma unroll
  for (int i = 0; i < 6; i++) q[i] = (_Float16)(e[i] / s);
#pragma unroll
  for (int i = 6; i < 32; i++) q[i] = (_Float16)0.f;
}

__global__ void concat_ctx(const float* __restrict__ slots, const float* __restrict__ a_enc,
                           _Float16* __restrict__ ctx16, int t, int total) {
  int i = blockIdx.x * blockDim.x + threadIdx.x;
  if (i >= total) return;
  int b = i / 384; int r = i - b * 384; int m = r >> 6; int d = r & 63;
  float v = (m < 5) ? slots[((long)b * 5 + m) * 64 + d]
                    : a_enc[((long)b * 5 + (t - 1)) * 64 + d];
  ctx16[i] = (_Float16)v;
}

__global__ void scatter_slots(const float* __restrict__ slots, float* __restrict__ out,
                              int Td, int t, int total) {
  int i = blockIdx.x * blockDim.x + threadIdx.x;
  if (i >= total) return;
  int b = i / 320; int r = i - b * 320;
  out[((long)b * Td + t) * 320 + r] = slots[i];
}

__global__ void bcast_pos_dec(const float* __restrict__ sl, long setStride,
                              const float* __restrict__ pw, const float* __restrict__ pb,
                              _Float16* __restrict__ o, int total) {
  int i = blockIdx.x * blockDim.x + threadIdx.x;
  if (i >= total) return;
  int d = i & 63; int p = (i >> 6) & 63; int m = i >> 12;
  int y = p >> 3, x = p & 7;
  int n = m / 5, k = m - n * 5;
  float g = 1.f / 7.f, gy = y * g, gx = x * g;
  float v = sl[(long)n * setStride + (long)k * 64 + d]
            + gy * pw[d] + gx * pw[64 + d] + (1.f - gy) * pw[128 + d] + (1.f - gx) * pw[192 + d] + pb[d];
  o[i] = (_Float16)v;
}

__global__ void combine_out(const float* __restrict__ d5, float* __restrict__ comb,
                            float* __restrict__ comp, float* __restrict__ mask,
                            int Td, int t, int total) {
  int i = blockIdx.x * blockDim.x + threadIdx.x;
  if (i >= total) return;
  int b = i >> 12, p = i & 4095;
  float a[5]; float m = -1e30f;
#pragma unroll
  for (int k = 0; k < 5; k++) {
    a[k] = d5[(((long)(b * 5 + k) * 4096) + p) * 4 + 3];
    m = fmaxf(m, a[k]);
  }
  float s = 0.f;
#pragma unroll
  for (int k = 0; k < 5; k++) { a[k] = expf(a[k] - m); s += a[k]; }
  float cb[3] = {0.f, 0.f, 0.f};
#pragma unroll
  for (int k = 0; k < 5; k++) {
    float mk = a[k] / s;
    mask[(((long)b * Td + t) * 5 + k) * 4096 + p] = mk;
#pragma unroll
    for (int c = 0; c < 3; c++) {
      float cv = d5[(((long)(b * 5 + k) * 4096) + p) * 4 + c];
      comp[((((long)b * Td + t) * 5 + k) * 4096 + p) * 3 + c] = cv;
      cb[c] += cv * mk;
    }
  }
#pragma unroll
  for (int c = 0; c < 3; c++)
    comb[(((long)b * Td + t) * 4096 + p) * 3 + c] = cb[c];
}

// ---------------------------------------------------------------------------
// Host orchestration
// ---------------------------------------------------------------------------
extern "C" void kernel_launch(void* const* d_in, const int* in_sizes, int n_in,
                              void* d_out, int out_size, void* d_ws, size_t ws_size,
                              hipStream_t stream) {
  (void)in_sizes; (void)n_in; (void)out_size; (void)ws_size;
  constexpr int B = 8, T = 6, H = 64, W = 64, K = 5, D = 64;
  constexpr int HW = H * W, BT = B * T, Tm1 = T - 1, M40 = B * K;

  const float* image      = (const float*)d_in[0];
  const float* action     = (const float*)d_in[1];
  const float* slots_init = (const float*)d_in[2];
  auto P = [&](int i) { return (const float*)d_in[3 + i]; };
  const float *enc_w1=P(0),*enc_b1=P(1),*enc_w2=P(2),*enc_b2=P(3),*enc_w3=P(4),*enc_b3=P(5),
              *enc_w4=P(6),*enc_b4=P(7),*enc_pos_w=P(8),*enc_pos_b=P(9),*enc_ln_s=P(10),*enc_ln_b=P(11),
              *enc_mw1=P(12),*enc_mb1=P(13),*enc_mw2=P(14),*enc_mb2=P(15),
              *sa_ln_in_s=P(16),*sa_ln_in_b=P(17),*sa_ln_sl_s=P(18),*sa_ln_sl_b=P(19),
              *sa_ln_mlp_s=P(20),*sa_ln_mlp_b=P(21),*sa_q=P(22),*sa_k=P(23),*sa_v=P(24),
              *gru_wx=P(25),*gru_wh=P(26),*gru_b=P(27),*sa_mw1=P(28),*sa_mb1=P(29),*sa_mw2=P(30),*sa_mb2=P(31),
              *act_w1=P(32),*act_b1=P(33),*act_w2=P(34),*act_b2=P(35),
              *dy_ln1_s=P(36),*dy_ln1_b=P(37),*dy_ln2_s=P(38),*dy_ln2_b=P(39),
              *dy_qw=P(40),*dy_qb=P(41),*dy_kw=P(42),*dy_kb=P(43),*dy_vw=P(44),*dy_vb=P(45),
              *dy_ow=P(46),*dy_ob=P(47),*dy_mw1=P(48),*dy_mb1=P(49),*dy_mw2=P(50),*dy_mb2=P(51),
              *dec_pos_w=P(52),*dec_pos_b=P(53),*dec_w1=P(54),*dec_b1=P(55),*dec_w2=P(56),*dec_b2=P(57),
              *dec_w3=P(58),*dec_b3=P(59),*dec_w4=P(60),*dec_b4=P(61),*dec_w5=P(62),*dec_b5=P(63);

  char* wsp = (char*)d_ws; size_t wso = 0;
  auto alloc = [&](size_t bytes) -> void* {
    void* r = wsp + wso; wso = (wso + bytes + 255) & ~(size_t)255; return r;
  };
  auto nblk = [](long n) { return dim3((unsigned)((n + 255) / 256)); };

  auto packW = [&](const float* src, int Kd, int N, int trans, int batch, long sstride) -> unsigned* {
    int ch = (Kd + 31) / 32, nt = (N + 15) / 16;
    unsigned* o = (unsigned*)alloc((size_t)batch * ch * nt * 256 * 4);
    pack_b<<<dim3(ch, nt, batch), 32, 0, stream>>>(src, sstride, Kd, N, trans, o, (long)ch * nt * 256);
    return o;
  };
  auto gemm = [&](const _Float16* A, int lda, long sA, const unsigned* Bp, long sB,
                  const float* bias, const float* add, long sAdd,
                  float* Cf, long sCf, _Float16* Ch, int ldch, long sCh,
                  int M, int N, int Kd, int batch, int act, float alpha) {
    wmma_gemm<<<dim3((M + 15) / 16, (N + 15) / 16, batch), 32, 0, stream>>>(
        A, lda, sA, Bp, sB, bias, add, sAdd, Cf, sCf, Ch, ldch, sCh, M, N, Kd, act, alpha);
  };

  // ---- pack all parameter matrices (deterministic, every call) -------------
  unsigned* Wp_e1  = packW(enc_w1, 75, 64, 0, 1, 0);
  unsigned* Wp_e2  = packW(enc_w2, 1600, 64, 0, 1, 0);
  unsigned* Wp_e3  = packW(enc_w3, 1600, 64, 0, 1, 0);
  unsigned* Wp_e4  = packW(enc_w4, 1600, 64, 0, 1, 0);
  unsigned* Wp_em1 = packW(enc_mw1, 64, 64, 0, 1, 0);
  unsigned* Wp_em2 = packW(enc_mw2, 64, 64, 0, 1, 0);
  unsigned* Wp_saq = packW(sa_q, 64, 64, 0, 1, 0);
  unsigned* Wp_sak = packW(sa_k, 64, 64, 0, 1, 0);
  unsigned* Wp_sav = packW(sa_v, 64, 64, 0, 1, 0);
  unsigned* Wp_gwx = packW(gru_wx, 64, 192, 0, 1, 0);
  unsigned* Wp_gwh = packW(gru_wh, 64, 192, 0, 1, 0);
  unsigned* Wp_sm1 = packW(sa_mw1, 64, 128, 0, 1, 0);
  unsigned* Wp_sm2 = packW(sa_mw2, 128, 64, 0, 1, 0);
  unsigned* Wp_aw1 = packW(act_w1, 6, 64, 0, 1, 0);
  unsigned* Wp_aw2 = packW(act_w2, 64, 64, 0, 1, 0);
  unsigned* Wp_dqw = packW(dy_qw, 64, 64, 0, 1, 0);
  unsigned* Wp_dkw = packW(dy_kw, 64, 64, 0, 1, 0);
  unsigned* Wp_dvw = packW(dy_vw, 64, 64, 0, 1, 0);
  unsigned* Wp_dow = packW(dy_ow, 64, 64, 0, 1, 0);
  unsigned* Wp_dm1 = packW(dy_mw1, 64, 256, 0, 1, 0);
  unsigned* Wp_dm2 = packW(dy_mw2, 256, 64, 0, 1, 0);
  unsigned* Wp_d1  = packW(dec_w1, 1600, 64, 0, 1, 0);
  unsigned* Wp_d2  = packW(dec_w2, 1600, 64, 0, 1, 0);
  unsigned* Wp_d3  = packW(dec_w3, 1600, 64, 0, 1, 0);
  unsigned* Wp_d4  = packW(dec_w4, 1600, 64, 0, 1, 0);
  unsigned* Wp_d5  = packW(dec_w5, 576, 4, 0, 1, 0);

  // ---- workspace buffers ---------------------------------------------------
  _Float16* img16 = (_Float16*)alloc((size_t)BT * HW * 3 * 2);
  _Float16* ha    = (_Float16*)alloc((size_t)BT * HW * D * 2);
  _Float16* hb    = (_Float16*)alloc((size_t)BT * HW * D * 2);
  float*    xfb   = (float*)alloc((size_t)BT * HW * D * 4);
  _Float16* xln16 = (_Float16*)alloc((size_t)BT * HW * D * 2);
  _Float16* g16   = (_Float16*)alloc((size_t)BT * HW * D * 2);
  float*    embed = (float*)alloc((size_t)BT * HW * D * 4);
  _Float16* act16 = (_Float16*)alloc((size_t)40 * 32 * 2);
  _Float16* ga16  = (_Float16*)alloc((size_t)40 * 64 * 2);
  float*    a_enc = (float*)alloc((size_t)40 * 64 * 4);
  float*    slots = (float*)alloc((size_t)40 * 64 * 4);
  _Float16* slots16 = (_Float16*)alloc((size_t)40 * 64 * 2);
  _Float16* inpn16  = (_Float16*)alloc((size_t)B * HW * D * 2);
  float*    kbuf  = (float*)alloc((size_t)B * HW * D * 4);
  float*    vbuf  = (float*)alloc((size_t)B * HW * D * 4);
  unsigned* kTp   = (unsigned*)alloc((size_t)B * 131072 * 4);
  unsigned* vP    = (unsigned*)alloc((size_t)B * 131072 * 4);
  _Float16* qin16 = (_Float16*)alloc((size_t)40 * 64 * 2);
  _Float16* q16   = (_Float16*)alloc((size_t)40 * 64 * 2);
  float*    logits = (float*)alloc((size_t)B * K * HW * 4);
  float*    attn  = (float*)alloc((size_t)B * K * HW * 4);
  float*    sums  = (float*)alloc((size_t)40 * 4);
  _Float16* attnA = (_Float16*)alloc((size_t)B * K * HW * 2);
  _Float16* upd16 = (_Float16*)alloc((size_t)40 * 64 * 2);
  float*    gxb   = (float*)alloc((size_t)40 * 192 * 4);
  float*    ghb   = (float*)alloc((size_t)40 * 192 * 4);
  _Float16* hres16 = (_Float16*)alloc((size_t)40 * 64 * 2);
  _Float16* m116  = (_Float16*)alloc((size_t)40 * 128 * 2);
  float*    prior = (float*)alloc((size_t)40 * 64 * 4);
  _Float16* prior16 = (_Float16*)alloc((size_t)40 * 64 * 2);
  _Float16* h16d  = (_Float16*)alloc((size_t)40 * 64 * 2);
  _Float16* ctx16 = (_Float16*)alloc((size_t)48 * 64 * 2);
  float*    kkb   = (float*)alloc((size_t)48 * 64 * 4);
  float*    vvb   = (float*)alloc((size_t)48 * 64 * 4);
  unsigned* kkTp  = (unsigned*)alloc((size_t)B * 512 * 4);
  unsigned* vvP   = (unsigned*)alloc((size_t)B * 1024 * 4);
  float*    attl  = (float*)alloc((size_t)240 * 4);
  _Float16* att16 = (_Float16*)alloc((size_t)40 * 32 * 2);
  _Float16* t116  = (_Float16*)alloc((size_t)40 * 64 * 2);
  _Float16* g216  = (_Float16*)alloc((size_t)40 * 256 * 2);
  _Float16* dx0   = (_Float16*)alloc((size_t)40 * 8 * 8 * 64 * 2);
  _Float16* dd1   = (_Float16*)alloc((size_t)40 * 16 * 16 * 64 * 2);
  _Float16* dd2   = (_Float16*)alloc((size_t)40 * 32 * 32 * 64 * 2);
  _Float16* dd3   = (_Float16*)alloc((size_t)40 * 64 * 64 * 64 * 2);
  _Float16* dd4   = (_Float16*)alloc((size_t)40 * 64 * 64 * 64 * 2);
  float*    dd5   = (float*)alloc((size_t)40 * 64 * 64 * 4 * 4);

  // ---- output regions ------------------------------------------------------
  float* out = (float*)d_out;
  float* o_pri    = out;                 // (B,5,K,D)
  float* o_pos    = out + 12800;         // (B,6,K,D)
  float* o_prcomb = out + 28160;
  float* o_prcomp = out + 519680;
  float* o_prmask = out + 2977280;
  float* o_pocomb = out + 3796480;
  float* o_pocomp = out + 4386304;
  float* o_pomask = out + 7335424;

  // ---- encoder -------------------------------------------------------------
  cvt_f32_f16<<<nblk((long)BT * HW * 3), 256, 0, stream>>>(image, img16, (long)BT * HW * 3);
  dim3 cg(BT * H * (W / 16));
  conv_wmma<4><<<cg, 32, 0, stream>>>(img16, Wp_e1, enc_b1, nullptr, ha, H, W, 3, 64, 5, 1);
  conv5_lds<<<dim3(BT * H), 128, 0, stream>>>(ha, Wp_e2, enc_b2, nullptr, hb, 1);
  conv5_lds<<<dim3(BT * H), 128, 0, stream>>>(hb, Wp_e3, enc_b3, nullptr, ha, 1);
  conv5_lds<<<dim3(BT * H), 128, 0, stream>>>(ha, Wp_e4, enc_b4, xfb, nullptr, 1);
  posadd_enc<<<nblk((long)BT * HW * D), 256, 0, stream>>>(xfb, enc_pos_w, enc_pos_b, (long)BT * HW * D);
  ln64<<<dim3(BT * HW, 1), 32, 0, stream>>>(xfb, 0, enc_ln_s, enc_ln_b, nullptr, xln16);
  gemm(xln16, 64, 0, Wp_em1, 0, enc_mb1, nullptr, 0, nullptr, 0, g16, 64, 0, BT * HW, 64, 64, 1, 1, 1.f);
  gemm(g16, 64, 0, Wp_em2, 0, enc_mb2, nullptr, 0, embed, 0, nullptr, 0, 0, BT * HW, 64, 64, 1, 0, 1.f);

  // ---- action encoder ------------------------------------------------------
  pad_cvt<<<nblk(40 * 32), 256, 0, stream>>>(action, act16, 40, 6, 32);
  gemm(act16, 32, 0, Wp_aw1, 0, act_b1, nullptr, 0, nullptr, 0, ga16, 64, 0, 40, 64, 6, 1, 1, 1.f);
  gemm(ga16, 64, 0, Wp_aw2, 0, act_b2, nullptr, 0, a_enc, 0, nullptr, 0, 0, 40, 64, 64, 1, 0, 1.f);

  // ---- slot attention (3 iters, operates on slots/slots16 in place) --------
  auto slot_attn = [&](const float* emb_t) {
    ln64<<<dim3(HW, B), 32, 0, stream>>>(emb_t, (long)T * HW * D, sa_ln_in_s, sa_ln_in_b, nullptr, inpn16);
    gemm(inpn16, 64, 0, Wp_sak, 0, nullptr, nullptr, 0, kbuf, 0, nullptr, 0, 0, B * HW, 64, 64, 1, 0, 1.f);
    gemm(inpn16, 64, 0, Wp_sav, 0, nullptr, nullptr, 0, vbuf, 0, nullptr, 0, 0, B * HW, 64, 64, 1, 0, 1.f);
    pack_b<<<dim3(2, 256, B), 32, 0, stream>>>(kbuf, (long)HW * 64, 64, HW, 1, kTp, 131072);
    pack_b<<<dim3(128, 4, B), 32, 0, stream>>>(vbuf, (long)HW * 64, HW, 64, 0, vP, 131072);
    for (int it = 0; it < 3; it++) {
      ln64<<<dim3(M40, 1), 32, 0, stream>>>(slots, 0, sa_ln_sl_s, sa_ln_sl_b, nullptr, qin16);
      gemm(qin16, 64, 0, Wp_saq, 0, nullptr, nullptr, 0, nullptr, 0, q16, 64, 0, M40, 64, 64, 1, 0, 1.f);
      gemm(q16, 64, (long)K * 64, kTp, 131072, nullptr, nullptr, 0,
           logits, (long)K * HW, nullptr, 0, 0, K, HW, 64, B, 0, 0.125f);
      colsoftmax5<<<nblk(B * HW), 256, 0, stream>>>(logits, attn, B * HW);
      rowsum<<<M40, 256, 0, stream>>>(attn, sums, HW);
      divnorm<<<nblk((long)B * K * HW), 256, 0, stream>>>(attn, sums, attnA, B * K * HW);
      gemm(attnA, HW, (long)K * HW, vP, 131072, nullptr, nullptr, 0,
           nullptr, 0, upd16, 64, (long)K * 64, K, 64, HW, B, 0, 1.f);
      gemm(upd16, 64, 0, Wp_gwx, 0, gru_b, nullptr, 0, gxb, 0, nullptr, 0, 0, M40, 192, 64, 1, 0, 1.f);
      gemm(slots16, 64, 0, Wp_gwh, 0, nullptr, nullptr, 0, ghb, 0, nullptr, 0, 0, M40, 192, 64, 1, 0, 1.f);
      gru_fuse<<<nblk(M40 * 64), 256, 0, stream>>>(gxb, ghb, slots, slots16, M40 * 64);
      ln64<<<dim3(M40, 1), 32, 0, stream>>>(slots, 0, sa_ln_mlp_s, sa_ln_mlp_b, nullptr, hres16);
      gemm(hres16, 64, 0, Wp_sm1, 0, sa_mb1, nullptr, 0, nullptr, 0, m116, 128, 0, M40, 128, 64, 1, 1, 1.f);
      gemm(m116, 128, 0, Wp_sm2, 0, sa_mb2, slots, 0, slots, 0, slots16, 64, 0, M40, 64, 128, 1, 0, 1.f);
    }
  };

  // ---- scan: post0 then (dynamics -> slot_attn) per step -------------------
  hipMemcpyAsync(slots, slots_init, (size_t)40 * 64 * 4, hipMemcpyDeviceToDevice, stream);
  cvt_f32_f16<<<nblk(40 * 64), 256, 0, stream>>>(slots, slots16, 40 * 64);
  slot_attn(embed);
  scatter_slots<<<nblk(40 * 64), 256, 0, stream>>>(slots, o_pos, T, 0, 40 * 64);

  for (int t = 1; t < T; t++) {
    concat_ctx<<<nblk(B * 6 * 64), 256, 0, stream>>>(slots, a_enc, ctx16, t, B * 6 * 64);
    hipMemcpyAsync(prior, slots, (size_t)40 * 64 * 4, hipMemcpyDeviceToDevice, stream);
    ln64<<<dim3(M40, 1), 32, 0, stream>>>(prior, 0, dy_ln1_s, dy_ln1_b, nullptr, h16d);
    gemm(h16d, 64, 0, Wp_dqw, 0, dy_qb, nullptr, 0, nullptr, 0, q16, 64, 0, M40, 64, 64, 1, 0, 1.f);
    gemm(ctx16, 64, 0, Wp_dkw, 0, dy_kb, nullptr, 0, kkb, 0, nullptr, 0, 0, B * 6, 64, 64, 1, 0, 1.f);
    gemm(ctx16, 64, 0, Wp_dvw, 0, dy_vb, nullptr, 0, vvb, 0, nullptr, 0, 0, B * 6, 64, 64, 1, 0, 1.f);
    pack_b<<<dim3(2, 1, B), 32, 0, stream>>>(kkb, (long)6 * 64, 64, 6, 1, kkTp, 512);
    pack_b<<<dim3(1, 4, B), 32, 0, stream>>>(vvb, (long)6 * 64, 6, 64, 0, vvP, 1024);
    gemm(q16, 64, (long)K * 64, kkTp, 512, nullptr, nullptr, 0,
         attl, 30, nullptr, 0, 0, K, 6, 64, B, 0, 0.125f);
    softmax6_cvt<<<nblk(M40), 256, 0, stream>>>(attl, att16, M40);
    gemm(att16, 32, (long)K * 32, vvP, 1024, nullptr, nullptr, 0,
         nullptr, 0, t116, 64, (long)K * 64, K, 64, 6, B, 0, 1.f);
    gemm(t116, 64, 0, Wp_dow, 0, dy_ob, prior, 0, prior, 0, nullptr, 0, 0, M40, 64, 64, 1, 0, 1.f);
    ln64<<<dim3(M40, 1), 32, 0, stream>>>(prior, 0, dy_ln2_s, dy_ln2_b, nullptr, h16d);
    gemm(h16d, 64, 0, Wp_dm1, 0, dy_mb1, nullptr, 0, nullptr, 0, g216, 256, 0, M40, 256, 64, 1, 2, 1.f);
    gemm(g216, 256, 0, Wp_dm2, 0, dy_mb2, prior, 0, prior, 0, prior16, 64, 0, M40, 64, 256, 1, 0, 1.f);
    scatter_slots<<<nblk(40 * 64), 256, 0, stream>>>(prior, o_pri, Tm1, t - 1, 40 * 64);
    hipMemcpyAsync(slots, prior, (size_t)40 * 64 * 4, hipMemcpyDeviceToDevice, stream);
    hipMemcpyAsync(slots16, prior16, (size_t)40 * 64 * 2, hipMemcpyDeviceToDevice, stream);
    slot_attn(embed + (long)t * HW * D);
    scatter_slots<<<nblk(40 * 64), 256, 0, stream>>>(slots, o_pos, T, t, 40 * 64);
  }

  // ---- decoder (per timestep: 40 maps of 8x8 -> 64x64) ---------------------
  auto decode = [&](const float* sl, long setStride, int Td, int t,
                    float* ocomb, float* ocomp, float* omask) {
    bcast_pos_dec<<<nblk((long)40 * 64 * 64), 256, 0, stream>>>(sl, setStride, dec_pos_w, dec_pos_b,
                                                                dx0, 40 * 64 * 64);
    deconv2_wmma<4><<<dim3(40 * 16 * 1), 32, 0, stream>>>(dx0, Wp_d1, dec_b1, dd1, 8, 8, 64, 64, 1);
    deconv2_wmma<4><<<dim3(40 * 32 * 2), 32, 0, stream>>>(dd1, Wp_d2, dec_b2, dd2, 16, 16, 64, 64, 1);
    deconv2_wmma<4><<<dim3(40 * 64 * 4), 32, 0, stream>>>(dd2, Wp_d3, dec_b3, dd3, 32, 32, 64, 64, 1);
    conv5_lds<<<dim3(40 * 64), 128, 0, stream>>>(dd3, Wp_d4, dec_b4, nullptr, dd4, 1);
    conv_wmma<1><<<dim3(40 * 64 * 4), 32, 0, stream>>>(dd4, Wp_d5, dec_b5, dd5, nullptr, 64, 64, 64, 4, 3, 0);
    combine_out<<<nblk(B * HW), 256, 0, stream>>>(dd5, ocomb, ocomp, omask, Td, t, B * HW);
  };
  for (int t = 0; t < Tm1; t++)
    decode(o_pri + (long)t * K * D, (long)Tm1 * K * D, Tm1, t, o_prcomb, o_prcomp, o_prmask);
  for (int t = 0; t < T; t++)
    decode(o_pos + (long)t * K * D, (long)T * K * D, T, t, o_pocomb, o_pocomp, o_pomask);
}
